// CVAE_85959475462881
// MI455X (gfx1250) — matrix operand order repaired and checked by reading, not compiled, not run on hardware
//
#include <hip/hip_runtime.h>
#include <math.h>

// ---------------- constants ----------------
#define Hd   1024
#define H4   4096
#define Vv   50257
#define Tt   100
#define LATn 32

// GEMM specialization: K = 1024, N <= 112 (7 x 16 N-tiles) for every call here.
#define KK     1024
#define NTILES 7
#define PHK    256                 // K-phase staged in LDS
#define BPAD   8                   // +16B row pad: kills 16-way DS bank conflicts
#define BSTR   (PHK + BPAD)        // LDS row stride in bf16 elements

typedef __attribute__((ext_vector_type(16))) __bf16 v16bf;
typedef __attribute__((ext_vector_type(8)))  float  v8f;
typedef __attribute__((ext_vector_type(8)))  float  f8;
typedef __attribute__((ext_vector_type(4)))  float  f4;

__device__ __forceinline__ float sigf(float x) { return 1.0f / (1.0f + expf(-x)); }

// ---------------- grid-wide barrier (persistent LSTM kernel) ----------------
__device__ __forceinline__ void grid_sync(int* flag, int* counter, int numWg) {
    __syncthreads();
    if (threadIdx.x == 0) {
        __threadfence();
        int prev = atomicAdd(counter, 1);
        if (prev == numWg - 1) {
            *counter = 0;   // reset for next barrier; ordered by release below
            __hip_atomic_store(flag, 1, __ATOMIC_RELEASE, __HIP_MEMORY_SCOPE_AGENT);
        } else {
            while (__hip_atomic_load(flag, __ATOMIC_ACQUIRE, __HIP_MEMORY_SCOPE_AGENT) == 0) {
                __builtin_amdgcn_s_sleep(1);
            }
        }
    }
    __syncthreads();
}

// ---------------- init: h0/c0 for encoder + barrier state ----------------
__global__ __launch_bounds__(256) void init_kernel(float* hbuf, float* cbuf,
                                                   const float* cond_emb,
                                                   const int* input_cond,
                                                   int* ibar) {
    int cidx = *input_cond;
    for (int j = threadIdx.x; j < Hd; j += 256) {
        cbuf[j] = 0.0f;
        hbuf[j] = (j < Hd - 8) ? 0.0f : cond_emb[cidx * 8 + (j - (Hd - 8))];
    }
    for (int j = threadIdx.x; j < 258; j += 256) ibar[j] = 0;
}

// ---------------- embedding gathers (enc: plain, dec: relu + SOS shift) ----
__global__ __launch_bounds__(256) void gather_kernel(const float* __restrict__ enc_emb,
                                                     const float* __restrict__ dec_emb,
                                                     const int* __restrict__ in_tok,
                                                     const int* __restrict__ tgt_tok,
                                                     float* __restrict__ Xenc,
                                                     float* __restrict__ Xdec) {
    int t = blockIdx.x;
    if (t < Tt) {
        int tok = in_tok[t];
        for (int k = threadIdx.x; k < Hd; k += 256)
            Xenc[t * Hd + k] = enc_emb[(size_t)tok * Hd + k];
    } else {
        int td = t - Tt;
        int tok = (td == 0) ? 0 : tgt_tok[td - 1];   // SOS = 0
        for (int k = threadIdx.x; k < Hd; k += 256) {
            float v = dec_emb[(size_t)tok * Hd + k];
            Xdec[td * Hd + k] = v > 0.0f ? v : 0.0f;
        }
    }
}

// ---------------- WMMA GEMM (NT): C[n*ldc+m] = sum_k A[m,k]*B[n,k] (+bias[m])
// A: [M][1024] fp32 row-major, streamed from HBM EXACTLY ONCE (each block owns
// a 128-row strip and all 7 N-tiles). B: [N][1024] fp32, L2-resident, staged
// into LDS as bf16 in 4 K-phases. fp32 accumulate, v_wmma_f32_16x16x32_bf16.
// 7 WMMAs per A-fragment; barrier-free inner loop.
__global__ __launch_bounds__(256) void gemm_nt_wmma(const float* __restrict__ A,
                                                    const float* __restrict__ B,
                                                    float* __restrict__ C,
                                                    const float* __restrict__ bias,
                                                    int M, int N, int ldc) {
    __shared__ __bf16 Bs[NTILES * 16][BSTR];       // 112 x 264 bf16 ~ 58KB
    const int tid  = threadIdx.x;
    const int wave = tid >> 5;
    const int lane = tid & 31;
    const int hi   = (lane >> 4) & 1;
    const int lo16 = lane & 15;
    const int m_base = blockIdx.x * 128 + wave * 16;
    const int mrow   = m_base + lo16;

    const bool avalid = mrow < M;
    const float* Ap = A + (size_t)mrow * KK;

    v8f acc[NTILES];
#pragma unroll
    for (int nt = 0; nt < NTILES; ++nt) acc[nt] = (v8f){};

    for (int kp = 0; kp < KK; kp += PHK) {
        // ---- stage B K-phase: 112 rows x 256 k, fp32 -> bf16 ----
        for (int e = tid * 4; e < NTILES * 16 * PHK; e += 256 * 4) {
            int n = e >> 8;          // e / PHK
            int k = e & (PHK - 1);
            f4 v = {};
            if (n < N) v = *(const f4*)(B + (size_t)n * KK + kp + k);
            Bs[n][k]     = (__bf16)v.x;
            Bs[n][k + 1] = (__bf16)v.y;
            Bs[n][k + 2] = (__bf16)v.z;
            Bs[n][k + 3] = (__bf16)v.w;
        }
        __syncthreads();

        // ---- barrier-free MAC phase: 8 x (1 A-frag, 7 WMMAs) ----
        for (int k0 = 0; k0 < PHK; k0 += 32) {
            // A fragment: lanes 0-15 hold K {0..7,16..23}, lanes 16-31 {8..15,24..31}
            v16bf af;
            if (avalid) {
                const float* ap = Ap + kp + k0 + hi * 8;
                f8 a0 = *(const f8*)(ap);
                f8 a1 = *(const f8*)(ap + 16);
                if (kp + k0 + 32 < KK) __builtin_prefetch(ap + 32, 0, 1);
#pragma unroll
                for (int j = 0; j < 8; ++j) { af[j] = (__bf16)a0[j]; af[8 + j] = (__bf16)a1[j]; }
            } else {
#pragma unroll
                for (int j = 0; j < 16; ++j) af[j] = (__bf16)0.0f;
            }

#pragma unroll
            for (int nt = 0; nt < NTILES; ++nt) {
                // B fragment: lane n = lo16; K {0..7,16..23} (hi=0) / {8..15,24..31} (hi=1)
                v16bf bf_;
                const __bf16* bsp = &Bs[nt * 16 + lo16][k0 + hi * 8];
#pragma unroll
                for (int j = 0; j < 8; ++j) { bf_[j] = bsp[j]; bf_[8 + j] = bsp[16 + j]; }
                acc[nt] = __builtin_amdgcn_wmma_f32_16x16x32_bf16(false, af, false, bf_,
                                                                  (short)0, acc[nt], false, false);
            }
        }
        __syncthreads();
    }

    // C/D layout: vgpr v, lane -> (m = m_base + v + 8*hi, n = nt*16 + lo16)
#pragma unroll
    for (int nt = 0; nt < NTILES; ++nt) {
#pragma unroll
        for (int v = 0; v < 8; ++v) {
            int m = m_base + v + hi * 8;
            int n = nt * 16 + lo16;
            if (m < M && n < N) {
                float r = acc[nt][v];
                if (bias) r += bias[m];
                C[(size_t)n * ldc + m] = r;
            }
        }
    }
}

// ---------------- persistent sequential LSTM --------------------------------
// Gates g = Gih[t] + Whh@h_prev + bih + bhh.  64 WGs x 256 threads; WG b owns
// h rows [16b,16b+16) and the 64 matching gate rows.  One grid barrier / step.
__global__ __launch_bounds__(256) void lstm_seq_kernel(const float* __restrict__ Gih,  // [T][4H]
                                                       const float* __restrict__ Whh,  // [4H][H]
                                                       const float* __restrict__ bih,
                                                       const float* __restrict__ bhh,
                                                       float* __restrict__ hbuf,       // [2][H]
                                                       float* __restrict__ cbuf,       // [2][H]
                                                       float* __restrict__ Hall,       // [T][H] or null
                                                       int T, int* counter, int* flags) {
    __shared__ float h_s[Hd];
    __shared__ float g_s[64];
    const int tid  = threadIdx.x;
    const int wave = tid >> 5;
    const int lane = tid & 31;
    const int j0   = blockIdx.x * 16;

    for (int t = 0; t < T; ++t) {
        const float* hp = hbuf + (size_t)(t & 1) * Hd;
        for (int k = tid; k < Hd; k += 256) h_s[k] = hp[k];
        __syncthreads();

        // 8 gate rows per wave; row_local = gate*16 + jj
#pragma unroll
        for (int it = 0; it < 8; ++it) {
            int row_local = wave * 8 + it;
            int gate = row_local >> 4;
            int jj   = row_local & 15;
            const float* wrow = Whh + (size_t)(gate * Hd + j0 + jj) * Hd;
            float sum = 0.0f;
#pragma unroll
            for (int kk = 0; kk < 32; ++kk) {
                int k = lane + kk * 32;
                sum += wrow[k] * h_s[k];
            }
            for (int off = 16; off > 0; off >>= 1) sum += __shfl_xor(sum, off, 32);
            if (lane == 0) g_s[row_local] = sum;
        }
        __syncthreads();

        if (tid < 16) {
            int j = j0 + tid;
            const float* gbase = Gih + (size_t)t * H4;
            float gi = gbase[j]            + g_s[tid]      + bih[j]            + bhh[j];
            float gf = gbase[Hd + j]       + g_s[16 + tid] + bih[Hd + j]       + bhh[Hd + j];
            float gg = gbase[2 * Hd + j]   + g_s[32 + tid] + bih[2 * Hd + j]   + bhh[2 * Hd + j];
            float go = gbase[3 * Hd + j]   + g_s[48 + tid] + bih[3 * Hd + j]   + bhh[3 * Hd + j];
            float cprev = cbuf[(size_t)(t & 1) * Hd + j];
            float c = sigf(gf) * cprev + sigf(gi) * tanhf(gg);
            float h = sigf(go) * tanhf(c);
            cbuf[(size_t)((t + 1) & 1) * Hd + j] = c;
            hbuf[(size_t)((t + 1) & 1) * Hd + j] = h;
            if (Hall) Hall[(size_t)t * Hd + j] = h;
        }
        grid_sync(&flags[t], counter, (int)gridDim.x);
    }
}

// ---------------- latent head: mean/logvar/reparam + lat2h/lat2c ------------
__global__ __launch_bounds__(256) void latent_kernel(const float* __restrict__ mean_W,
                                                     const float* __restrict__ mean_b,
                                                     const float* __restrict__ logvar_W,
                                                     const float* __restrict__ logvar_b,
                                                     const float* __restrict__ lat2h_W,
                                                     const float* __restrict__ lat2h_b,
                                                     const float* __restrict__ lat2c_W,
                                                     const float* __restrict__ lat2c_b,
                                                     const float* __restrict__ cond_emb,
                                                     const float* __restrict__ eps,
                                                     const int* __restrict__ tgt_cond,
                                                     float* __restrict__ out_mean,
                                                     float* __restrict__ out_logvar,
                                                     float* __restrict__ hbuf,   // slot 0: in h_enc, out hd
                                                     float* __restrict__ cbuf) { // slot 0: out cd
    __shared__ float h_s[Hd];
    __shared__ float ml[64];
    __shared__ float zc[LATn + 8];
    const int tid  = threadIdx.x;
    const int wave = tid >> 5;
    const int lane = tid & 31;

    for (int k = tid; k < Hd; k += 256) h_s[k] = hbuf[k];
    __syncthreads();

#pragma unroll
    for (int it = 0; it < 8; ++it) {
        int r = wave * 8 + it;                        // 0..31 mean, 32..63 logvar
        const float* wrow = (r < LATn) ? (mean_W + (size_t)r * Hd)
                                       : (logvar_W + (size_t)(r - LATn) * Hd);
        float sum = 0.0f;
#pragma unroll
        for (int kk = 0; kk < 32; ++kk) {
            int k = lane + kk * 32;
            sum += wrow[k] * h_s[k];
        }
        for (int off = 16; off > 0; off >>= 1) sum += __shfl_xor(sum, off, 32);
        if (lane == 0) ml[r] = sum;
    }
    __syncthreads();

    if (tid < LATn) {
        float m  = ml[tid] + mean_b[tid];
        float lv = ml[LATn + tid] + logvar_b[tid];
        out_mean[tid]   = m;
        out_logvar[tid] = lv;
        zc[tid] = eps[tid] * expf(0.5f * lv) + m;
    } else if (tid < LATn + 8) {
        zc[tid] = cond_emb[(*tgt_cond) * 8 + (tid - LATn)];
    }
    __syncthreads();

    for (int r = tid; r < 2 * Hd; r += 256) {
        bool is_h = r < Hd;
        int  rr   = is_h ? r : r - Hd;
        const float* wrow = (is_h ? lat2h_W : lat2c_W) + (size_t)rr * (LATn + 8);
        float s = is_h ? lat2h_b[rr] : lat2c_b[rr];
#pragma unroll
        for (int j = 0; j < LATn + 8; ++j) s += wrow[j] * zc[j];
        if (is_h) hbuf[rr] = s; else cbuf[rr] = s;
    }
}

// ---------------- host launcher ---------------------------------------------
extern "C" void kernel_launch(void* const* d_in, const int* in_sizes, int n_in,
                              void* d_out, int out_size, void* d_ws, size_t ws_size,
                              hipStream_t stream) {
    const float* cond_emb  = (const float*)d_in[0];
    const float* enc_emb   = (const float*)d_in[1];
    const float* enc_Wih   = (const float*)d_in[2];
    const float* enc_Whh   = (const float*)d_in[3];
    const float* enc_bih   = (const float*)d_in[4];
    const float* enc_bhh   = (const float*)d_in[5];
    const float* mean_W    = (const float*)d_in[6];
    const float* mean_b    = (const float*)d_in[7];
    const float* logvar_W  = (const float*)d_in[8];
    const float* logvar_b  = (const float*)d_in[9];
    const float* lat2h_W   = (const float*)d_in[10];
    const float* lat2h_b   = (const float*)d_in[11];
    const float* lat2c_W   = (const float*)d_in[12];
    const float* lat2c_b   = (const float*)d_in[13];
    const float* dec_emb   = (const float*)d_in[14];
    const float* dec_Wih   = (const float*)d_in[15];
    const float* dec_Whh   = (const float*)d_in[16];
    const float* dec_bih   = (const float*)d_in[17];
    const float* dec_bhh   = (const float*)d_in[18];
    const float* out_W     = (const float*)d_in[19];
    const float* out_b     = (const float*)d_in[20];
    const float* eps       = (const float*)d_in[21];
    const int*   in_tok    = (const int*)d_in[22];
    const int*   in_cond   = (const int*)d_in[23];
    const int*   tgt_tok   = (const int*)d_in[24];
    const int*   tgt_cond  = (const int*)d_in[25];

    float* out = (float*)d_out;
    float* out_logits = out;                   // [T][V]
    float* out_mean   = out + (size_t)Tt * Vv; // [32]
    float* out_logvar = out_mean + LATn;       // [32]

    // workspace layout (fp32)
    float* ws   = (float*)d_ws;
    float* Xenc = ws;                          // [T][H]
    float* Xdec = Xenc + Tt * Hd;              // [T][H]
    float* GihE = Xdec + Tt * Hd;              // [T][4H]
    float* GihD = GihE + Tt * H4;              // [T][4H]
    float* Hall = GihD + Tt * H4;              // [T][H]
    float* hbuf = Hall + Tt * Hd;              // [2][H]
    float* cbuf = hbuf + 2 * Hd;               // [2][H]
    int*   ibar = (int*)(cbuf + 2 * Hd);       // flags[0..127] enc, [128..255] dec, [256..257] counters

    const int NWG_LSTM = 64;

    // 1. init barrier state + encoder h0/c0
    init_kernel<<<1, 256, 0, stream>>>(hbuf, cbuf, cond_emb, in_cond, ibar);

    // 2. embedding gathers (enc plain, dec relu with SOS shift)
    gather_kernel<<<2 * Tt, 256, 0, stream>>>(enc_emb, dec_emb, in_tok, tgt_tok, Xenc, Xdec);

    // 3. precompute input-to-hidden gate GEMMs (WMMA): Gih[t][m] = Wih @ x_t
    gemm_nt_wmma<<<H4 / 128, 256, 0, stream>>>(enc_Wih, Xenc, GihE, nullptr, H4, Tt, H4);
    gemm_nt_wmma<<<H4 / 128, 256, 0, stream>>>(dec_Wih, Xdec, GihD, nullptr, H4, Tt, H4);

    // 4. encoder recurrence (persistent, one grid barrier per step)
    lstm_seq_kernel<<<NWG_LSTM, 256, 0, stream>>>(GihE, enc_Whh, enc_bih, enc_bhh,
                                                  hbuf, cbuf, nullptr, Tt,
                                                  &ibar[256], &ibar[0]);

    // 5. latent head -> mean/logvar to output, hd/cd into ping-pong slot 0
    latent_kernel<<<1, 256, 0, stream>>>(mean_W, mean_b, logvar_W, logvar_b,
                                         lat2h_W, lat2h_b, lat2c_W, lat2c_b,
                                         cond_emb, eps, tgt_cond,
                                         out_mean, out_logvar, hbuf, cbuf);

    // 6. decoder recurrence, storing every h_t
    lstm_seq_kernel<<<NWG_LSTM, 256, 0, stream>>>(GihD, dec_Whh, dec_bih, dec_bhh,
                                                  hbuf, cbuf, Hall, Tt,
                                                  &ibar[257], &ibar[128]);

    // 7. big output projection as one WMMA GEMM: logits[t][v] = out_W @ h_t + out_b
    //    out_W (206 MB) streamed from HBM exactly once.
    gemm_nt_wmma<<<(Vv + 127) / 128, 256, 0, stream>>>(out_W, Hall, out_logits, out_b,
                                                       Vv, Tt, Vv);
}
